// GATDRL_56822417326669
// MI455X (gfx1250) — matrix operand order repaired and checked by reading, not compiled, not run on hardware
//
#include <hip/hip_runtime.h>
#include <hip/hip_bf16.h>

// ---------------------------------------------------------------------------
// GAT-DRL fused pipeline for gfx1250 (MI455X)
// wave32 + v_wmma_f32_16x16x32_bf16 + TDM (tensor_load_to_lds) double buffering
// ---------------------------------------------------------------------------

#define N_NODES 4096
#define S_DIM_C 64
#define HID_C   128
#define HEADS_C 4
#define F_C     32
#define A_DIM_C 32
#define ALPHA_C 0.2f
#define NEG_C   (-9000000000000000.0f)

typedef __attribute__((ext_vector_type(16))) __bf16 v16bf;
typedef __attribute__((ext_vector_type(8)))  float  v8f;
typedef unsigned int u32x4 __attribute__((ext_vector_type(4)));
typedef int          i32x8 __attribute__((ext_vector_type(8)));
typedef int          i32x4 __attribute__((ext_vector_type(4)));

union V16U { v16bf v; unsigned short u[16]; };

#if defined(__gfx1250__) && __has_builtin(__builtin_amdgcn_tensor_load_to_lds) && \
    __has_builtin(__builtin_amdgcn_s_wait_tensorcnt)
#define USE_TDM 1
#else
#define USE_TDM 0
#endif

__device__ __forceinline__ unsigned short f2bf(float f) {
    unsigned int u = __float_as_uint(f);
    unsigned int r = u + 0x7FFFu + ((u >> 16) & 1u);   // round-to-nearest-even
    return (unsigned short)(r >> 16);
}
__device__ __forceinline__ float bf2f(unsigned short s) {
    return __uint_as_float(((unsigned int)s) << 16);
}

__device__ __forceinline__ v8f wmma_bf16(v16bf a, v16bf b, v8f c) {
    return __builtin_amdgcn_wmma_f32_16x16x32_bf16(
        false, a, false, b, (short)0, c, false, false);
}

// A-frag (16x32, 16-bit): lane L -> row L%16; elems 0..7 -> k = k0 + half*8 + i,
// elems 8..15 -> k = k0 + 16 + half*8 + i   (half = L>=16)
__device__ __forceinline__ v16bf load_A_f32(const float* src, int ld, int row0,
                                            int k0, int lane) {
    int m  = row0 + (lane & 15);
    int kb = k0 + ((lane & 16) ? 8 : 0);
    V16U r;
#pragma unroll
    for (int i = 0; i < 8; ++i) {
        r.u[i]     = f2bf(src[m * ld + kb + i]);
        r.u[i + 8] = f2bf(src[m * ld + kb + 16 + i]);
    }
    return r.v;
}

__device__ __forceinline__ v16bf load_A_lds(const unsigned short* src, int ld,
                                            int k0, int lane) {
    int m  = lane & 15;
    int kb = k0 + ((lane & 16) ? 8 : 0);
    V16U r;
#pragma unroll
    for (int i = 0; i < 8; ++i) {
        r.u[i]     = src[m * ld + kb + i];
        r.u[i + 8] = src[m * ld + kb + 16 + i];
    }
    return r.v;
}

// Pre-packed B fragments: [tile][lane][16] contiguous bf16 -> two b128 loads
__device__ __forceinline__ v16bf load_B_pk(const unsigned short* base, int tile,
                                           int lane) {
    return *(const v16bf*)(base + (((size_t)tile * 32 + lane) << 4));
}

// ---------------------------------------------------------------------------
// K0: pack an f32 KxN weight matrix into B-fragment-ordered bf16
// tile = kt*(N/16)+nt ; element (lane,i): k = kt*32 + (lane&16?8:0) + (i<8 ? i : i+8),
// n = nt*16 + lane&15
// ---------------------------------------------------------------------------
__global__ void GATDRL_k0_pack(const float* __restrict__ src,
                               unsigned short* __restrict__ dst,
                               int K, int N) {
    int tid = blockIdx.x * blockDim.x + threadIdx.x;
    if (tid >= K * N) return;
    int tile = tid >> 9;
    int lane = (tid >> 4) & 31;
    int i    = tid & 15;
    int ntn  = N >> 4;
    int kt   = tile / ntn, nt = tile % ntn;
    int k = kt * 32 + ((lane & 16) ? 8 : 0) + (i < 8 ? i : i + 8);
    int n = nt * 16 + (lane & 15);
    dst[tid] = f2bf(src[k * N + n]);
}

// ---------------------------------------------------------------------------
// K1: x = relu(relu(state@W1+b1)@W2+b2);  h[hh] = x @ Wg[hh]  (stored bf16)
// one wave per 16-node tile; packed bf16 weights
// ---------------------------------------------------------------------------
__global__ void GATDRL_k1_encode(const float* __restrict__ state,
                                 const unsigned short* __restrict__ W1f,
                                 const float* __restrict__ b1,
                                 const unsigned short* __restrict__ W2f,
                                 const float* __restrict__ b2,
                                 const unsigned short* __restrict__ Wgf,
                                 unsigned short* __restrict__ h_bf16) {
    __shared__ unsigned short xl[16 * HID_C];
    const int lane = threadIdx.x;
    const int row0 = blockIdx.x * 16;
    const int n    = lane & 15;
    const int rb   = (lane & 16) ? 8 : 0;

    // ---- layer 1: (16x64) @ (64x128) ----
    v16bf A0 = load_A_f32(state, S_DIM_C, row0, 0, lane);
    v16bf A1 = load_A_f32(state, S_DIM_C, row0, 32, lane);
#pragma unroll
    for (int nt = 0; nt < 8; ++nt) {
        v8f c = {};
        c = wmma_bf16(A0, load_B_pk(W1f, 0 * 8 + nt, lane), c);
        c = wmma_bf16(A1, load_B_pk(W1f, 1 * 8 + nt, lane), c);
        float bias = b1[nt * 16 + n];
#pragma unroll
        for (int r = 0; r < 8; ++r) {
            float v = c[r] + bias;
            v = v > 0.f ? v : 0.f;
            xl[(rb + r) * HID_C + nt * 16 + n] = f2bf(v);
        }
    }
    __syncthreads();

    v16bf Ax[4];
#pragma unroll
    for (int kt = 0; kt < 4; ++kt) Ax[kt] = load_A_lds(xl, HID_C, kt * 32, lane);
    __syncthreads();

    // ---- layer 2: (16x128) @ (128x128) ----
#pragma unroll
    for (int nt = 0; nt < 8; ++nt) {
        v8f c = {};
#pragma unroll
        for (int kt = 0; kt < 4; ++kt)
            c = wmma_bf16(Ax[kt], load_B_pk(W2f, kt * 8 + nt, lane), c);
        float bias = b2[nt * 16 + n];
#pragma unroll
        for (int r = 0; r < 8; ++r) {
            float v = c[r] + bias;
            v = v > 0.f ? v : 0.f;
            xl[(rb + r) * HID_C + nt * 16 + n] = f2bf(v);
        }
    }
    __syncthreads();
#pragma unroll
    for (int kt = 0; kt < 4; ++kt) Ax[kt] = load_A_lds(xl, HID_C, kt * 32, lane);

    // ---- per-head projection: h[hh] = x @ Wg[hh] (128x32, 8 tiles/head) ----
    for (int hh = 0; hh < HEADS_C; ++hh) {
        const unsigned short* Wgh = Wgf + hh * 8 * 512;
#pragma unroll
        for (int nt = 0; nt < 2; ++nt) {
            v8f c = {};
#pragma unroll
            for (int kt = 0; kt < 4; ++kt)
                c = wmma_bf16(Ax[kt], load_B_pk(Wgh, kt * 2 + nt, lane), c);
            int fcol = nt * 16 + n;
#pragma unroll
            for (int r = 0; r < 8; ++r)
                h_bf16[((size_t)hh * N_NODES + row0 + rb + r) * F_C + fcol] = f2bf(c[r]);
        }
    }
}

// ---------------------------------------------------------------------------
// K1b: attention scores s1[h,n] = h.a1, s2[h,n] = h.a2
// ---------------------------------------------------------------------------
__global__ void GATDRL_k1b_scores(const unsigned short* __restrict__ h_bf16,
                                  const float* __restrict__ ag,
                                  float* __restrict__ s1,
                                  float* __restrict__ s2) {
    int idx = blockIdx.x * blockDim.x + threadIdx.x;
    if (idx >= HEADS_C * N_NODES) return;
    int hh = idx / N_NODES;
    const unsigned short* hp = h_bf16 + (size_t)idx * F_C;
    const float* a1 = ag + hh * 2 * F_C;
    const float* a2 = a1 + F_C;
    float acc1 = 0.f, acc2 = 0.f;
#pragma unroll
    for (int f = 0; f < F_C; ++f) {
        float v = bf2f(hp[f]);
        acc1 += v * a1[f];
        acc2 += v * a2[f];
    }
    s1[idx] = acc1;
    s2[idx] = acc2;
}

// ---------------------------------------------------------------------------
// TDM: issue a 16x32 int32 adj tile load into LDS via the Tensor Data Mover
// D# per CDNA5 ISA (2-D, data_size=4B, type=2)
// 6-arg builtin form (this toolchain): (g0, g1, g2, g3, g4, cpol)
// ---------------------------------------------------------------------------
#if USE_TDM
__device__ __forceinline__ void tdm_load_adj(const int* adj, unsigned lds_off,
                                             int row0, int m0) {
    unsigned long long ga =
        (unsigned long long)(uintptr_t)(adj + (size_t)row0 * N_NODES + m0);
    u32x4 g0;
    g0[0] = 1u;                                            // count=1 (valid D#)
    g0[1] = lds_off;                                       // lds_addr
    g0[2] = (unsigned)ga;                                  // global_addr[31:0]
    g0[3] = (unsigned)((ga >> 32) & 0x01FFFFFFull)         // global_addr[56:32]
            | 0x80000000u;                                 // type=2 ("image")
    i32x8 g1;
    g1[0] = 2 << 16;                                       // data_size=4B
    g1[1] = (int)((unsigned)N_NODES << 16);                // tensor_dim0 lo16
    g1[2] = (int)((unsigned)N_NODES << 16);                // dim0 hi | tensor_dim1 lo16
    g1[3] = 32 << 16;                                      // dim1 hi | tile_dim0=32
    g1[4] = 16;                                            // tile_dim1=16, tile_dim2=0
    g1[5] = N_NODES;                                       // tensor_dim0_stride[31:0]
    g1[6] = 0;                                             // stride0 hi | stride1 lo
    g1[7] = 0;                                             // stride1 hi
    i32x4 z4 = {0, 0, 0, 0};
    i32x8 z8 = {0, 0, 0, 0, 0, 0, 0, 0};
    __builtin_amdgcn_tensor_load_to_lds(g0, g1, z4, z4, z8, 0);
}
#endif

// ---------------------------------------------------------------------------
// K2: fused masked-softmax attention (flash-style). Block = 4 waves (wave==head),
// one 16-query tile; adj tile streamed ONCE via TDM, double-buffered in LDS,
// shared across the 4 heads.   out -> feat[n, head*F + f] (f32)
// ---------------------------------------------------------------------------
__global__ void GATDRL_k2_attn(const int* __restrict__ adj,
                               const unsigned short* __restrict__ h_bf16,
                               const float* __restrict__ s1,
                               const float* __restrict__ s2,
                               float* __restrict__ feat) {
    __shared__ int adjl[2][16 * 32];
    const int tid  = threadIdx.x;
    const int lane = tid & 31;
    const int hh   = tid >> 5;
    const int row0 = blockIdx.x * 16;
    const int n    = lane & 15;
    const int half = (lane & 16) ? 1 : 0;
    const int rb   = half * 8;

    int kidx[16];
#pragma unroll
    for (int i = 0; i < 8; ++i) {
        kidx[i]     = half * 16 + i;
        kidx[i + 8] = half * 16 + 8 + i;
    }

    const float s1q = s1[hh * N_NODES + row0 + n];
    float M = -__builtin_inff();
    float S = 0.f;
    v8f O0 = {}, O1 = {};

#if USE_TDM
    const unsigned lds_base = (unsigned)(uintptr_t)(&adjl[0][0]);
    if (hh == 0) tdm_load_adj(adj, lds_base, row0, 0);
#endif

    for (int m0 = 0; m0 < N_NODES; m0 += 32) {
#if USE_TDM
        const int buf = (m0 >> 5) & 1;
        if (hh == 0) {
            if (m0 + 32 < N_NODES) {
                tdm_load_adj(adj, lds_base + (unsigned)((buf ^ 1) << 11), row0, m0 + 32);
                __builtin_amdgcn_s_wait_tensorcnt(1);   // current tile landed
            } else {
                __builtin_amdgcn_s_wait_tensorcnt(0);
            }
        }
        __syncthreads();
        const int* adjt = adjl[buf];
#else
#pragma unroll
        for (int j = 0; j < 4; ++j) {
            int l = tid * 4 + j;
            adjl[0][l] = adj[(size_t)(row0 + (l >> 5)) * N_NODES + m0 + (l & 31)];
        }
        __syncthreads();
        const int* adjt = adjl[0];
#endif

        const float* s2h = s2 + hh * N_NODES + m0;
        float e[16];
        float lmax = -__builtin_inff();
#pragma unroll
        for (int i = 0; i < 16; ++i) {
            float t = s1q + s2h[kidx[i]];
            t = (t > 0.f) ? t : ALPHA_C * t;                 // leaky relu
            t = (adjt[n * 32 + kidx[i]] > 0) ? t : NEG_C;    // mask
            e[i] = t;
            lmax = fmaxf(lmax, t);
        }
        // row data lives in lanes {n, n+16} -> single xor-16 combine
        float rmax = fmaxf(lmax, __shfl_xor(lmax, 16, 32));
        float Mn   = fmaxf(M, rmax);
        float sc   = __expf(M - Mn);
        M = Mn;

        float lsum = 0.f;
        V16U P;
#pragma unroll
        for (int i = 0; i < 16; ++i) {
            float p = __expf(e[i] - Mn);
            lsum += p;
            P.u[i] = f2bf(p);
        }
        S = S * sc + (lsum + __shfl_xor(lsum, 16, 32));

#pragma unroll
        for (int r = 0; r < 8; ++r) {
            float scr = __shfl(sc, rb + r, 32);
            O0[r] *= scr;
            O1[r] *= scr;
        }

#pragma unroll
        for (int ft = 0; ft < 2; ++ft) {
            V16U B;
            const unsigned short* hp =
                h_bf16 + ((size_t)hh * N_NODES + m0 + half * 16) * F_C + ft * 16 + n;
#pragma unroll
            for (int i = 0; i < 16; ++i) B.u[i] = hp[i * F_C];
            if (ft == 0) O0 = wmma_bf16(P.v, B.v, O0);
            else         O1 = wmma_bf16(P.v, B.v, O1);
        }
        __syncthreads();
    }

    float inv = 1.f / S;
#pragma unroll
    for (int r = 0; r < 8; ++r) {
        float invr = __shfl(inv, rb + r, 32);
        feat[(size_t)(row0 + rb + r) * HID_C + hh * F_C + n]      = O0[r] * invr;
        feat[(size_t)(row0 + rb + r) * HID_C + hh * F_C + 16 + n] = O1[r] * invr;
    }
}

// ---------------------------------------------------------------------------
// K3: policy head (softmax over 32 logits) + value head. one wave / 16 rows
// ---------------------------------------------------------------------------
__global__ void GATDRL_k3_heads(const float* __restrict__ feat,
                                const unsigned short* __restrict__ Wp1f,
                                const float* __restrict__ bp1,
                                const unsigned short* __restrict__ Wp2f,
                                const float* __restrict__ bp2,
                                const unsigned short* __restrict__ Wv1f,
                                const float* __restrict__ bv1,
                                const float* __restrict__ Wv2,
                                const float* __restrict__ bv2,
                                float* __restrict__ probs,
                                float* __restrict__ value) {
    __shared__ unsigned short xl[16 * HID_C];
    const int lane = threadIdx.x;
    const int row0 = blockIdx.x * 16;
    const int n    = lane & 15;
    const int half = (lane & 16) ? 1 : 0;
    const int rb   = half * 8;

    v16bf Af[4];
#pragma unroll
    for (int kt = 0; kt < 4; ++kt)
        Af[kt] = load_A_f32(feat, HID_C, row0, kt * 32, lane);

    // ---- policy hidden ----
#pragma unroll
    for (int nt = 0; nt < 8; ++nt) {
        v8f c = {};
#pragma unroll
        for (int kt = 0; kt < 4; ++kt)
            c = wmma_bf16(Af[kt], load_B_pk(Wp1f, kt * 8 + nt, lane), c);
        float bias = bp1[nt * 16 + n];
#pragma unroll
        for (int r = 0; r < 8; ++r) {
            float v = c[r] + bias;
            v = v > 0.f ? v : 0.f;
            xl[(rb + r) * HID_C + nt * 16 + n] = f2bf(v);
        }
    }
    __syncthreads();
    v16bf Ah[4];
#pragma unroll
    for (int kt = 0; kt < 4; ++kt) Ah[kt] = load_A_lds(xl, HID_C, kt * 32, lane);

    // ---- logits (16x128 @ 128x32, tiles kt*2+nt) + row softmax over 32 ----
    v8f L0 = {}, L1 = {};
#pragma unroll
    for (int kt = 0; kt < 4; ++kt) {
        L0 = wmma_bf16(Ah[kt], load_B_pk(Wp2f, kt * 2 + 0, lane), L0);
        L1 = wmma_bf16(Ah[kt], load_B_pk(Wp2f, kt * 2 + 1, lane), L1);
    }
    float bb0 = bp2[n], bb1 = bp2[16 + n];
#pragma unroll
    for (int r = 0; r < 8; ++r) {
        float v0 = L0[r] + bb0, v1 = L1[r] + bb1;
        float mx = fmaxf(v0, v1);
#pragma unroll
        for (int o = 1; o < 16; o <<= 1) mx = fmaxf(mx, __shfl_xor(mx, o, 32));
        float p0 = __expf(v0 - mx), p1 = __expf(v1 - mx);
        float sm = p0 + p1;
#pragma unroll
        for (int o = 1; o < 16; o <<= 1) sm += __shfl_xor(sm, o, 32);
        float invs = 1.f / sm;
        probs[(size_t)(row0 + rb + r) * A_DIM_C + n]      = p0 * invs;
        probs[(size_t)(row0 + rb + r) * A_DIM_C + 16 + n] = p1 * invs;
    }
    __syncthreads();

    // ---- value hidden (reuse LDS) ----
#pragma unroll
    for (int nt = 0; nt < 8; ++nt) {
        v8f c = {};
#pragma unroll
        for (int kt = 0; kt < 4; ++kt)
            c = wmma_bf16(Af[kt], load_B_pk(Wv1f, kt * 8 + nt, lane), c);
        float bias = bv1[nt * 16 + n];
#pragma unroll
        for (int r = 0; r < 8; ++r) {
            float v = c[r] + bias;
            v = v > 0.f ? v : 0.f;
            xl[(rb + r) * HID_C + nt * 16 + n] = f2bf(v);
        }
    }
    __syncthreads();

    // ---- value = hidden @ Wv2 + bv2 : row n, lanes {n, n+16} split columns ----
    {
        float part = 0.f;
#pragma unroll
        for (int c = 0; c < 64; ++c) {
            int col = half * 64 + c;
            part += bf2f(xl[n * HID_C + col]) * Wv2[col];
        }
        float tot = part + __shfl_xor(part, 16, 32);
        if (half == 0) value[row0 + n] = tot + bv2[0];
    }
}

// ---------------------------------------------------------------------------
extern "C" void kernel_launch(void* const* d_in, const int* in_sizes, int n_in,
                              void* d_out, int out_size, void* d_ws, size_t ws_size,
                              hipStream_t stream) {
    (void)in_sizes; (void)n_in; (void)out_size; (void)ws_size;
    const float* state = (const float*)d_in[0];
    const int*   adj   = (const int*)d_in[1];
    const float* W1    = (const float*)d_in[2];
    const float* b1    = (const float*)d_in[3];
    const float* W2    = (const float*)d_in[4];
    const float* b2    = (const float*)d_in[5];
    const float* Wg    = (const float*)d_in[6];
    const float* ag    = (const float*)d_in[7];
    const float* Wp1   = (const float*)d_in[8];
    const float* bp1   = (const float*)d_in[9];
    const float* Wp2   = (const float*)d_in[10];
    const float* bp2   = (const float*)d_in[11];
    const float* Wv1   = (const float*)d_in[12];
    const float* bv1   = (const float*)d_in[13];
    const float* Wv2   = (const float*)d_in[14];
    const float* bv2   = (const float*)d_in[15];

    // ---- workspace layout ----
    unsigned short* h_bf16 = (unsigned short*)d_ws;                    // 1 MB
    float* s1   = (float*)((char*)d_ws + (1 << 20));                   // 64 KB
    float* s2   = s1 + HEADS_C * N_NODES;                              // 64 KB
    float* feat = s2 + HEADS_C * N_NODES;                              // 2 MB
    unsigned short* W1f  = (unsigned short*)(feat + (size_t)N_NODES * HID_C);
    unsigned short* W2f  = W1f + 8192;
    unsigned short* Wgf  = W2f + 16384;
    unsigned short* Wp1f = Wgf + 16384;
    unsigned short* Wp2f = Wp1f + 16384;
    unsigned short* Wv1f = Wp2f + 4096;

    float* probs = (float*)d_out;                                      // 4096*32
    float* value = probs + (size_t)N_NODES * A_DIM_C;                  // 4096

    // ---- pack weights into B-fragment bf16 (one-shot, L2-resident) ----
    GATDRL_k0_pack<<<(S_DIM_C * HID_C + 255) / 256, 256, 0, stream>>>(W1, W1f, S_DIM_C, HID_C);
    GATDRL_k0_pack<<<(HID_C * HID_C + 255) / 256, 256, 0, stream>>>(W2, W2f, HID_C, HID_C);
    for (int hh = 0; hh < HEADS_C; ++hh)
        GATDRL_k0_pack<<<(HID_C * F_C + 255) / 256, 256, 0, stream>>>(
            Wg + (size_t)hh * HID_C * F_C, Wgf + (size_t)hh * HID_C * F_C, HID_C, F_C);
    GATDRL_k0_pack<<<(HID_C * HID_C + 255) / 256, 256, 0, stream>>>(Wp1, Wp1f, HID_C, HID_C);
    GATDRL_k0_pack<<<(HID_C * A_DIM_C + 255) / 256, 256, 0, stream>>>(Wp2, Wp2f, HID_C, A_DIM_C);
    GATDRL_k0_pack<<<(HID_C * HID_C + 255) / 256, 256, 0, stream>>>(Wv1, Wv1f, HID_C, HID_C);

    GATDRL_k1_encode<<<N_NODES / 16, 32, 0, stream>>>(state, W1f, b1, W2f, b2, Wgf, h_bf16);
    GATDRL_k1b_scores<<<(HEADS_C * N_NODES + 255) / 256, 256, 0, stream>>>(h_bf16, ag, s1, s2);
    GATDRL_k2_attn<<<N_NODES / 16, 128, 0, stream>>>(adj, h_bf16, s1, s2, feat);
    GATDRL_k3_heads<<<N_NODES / 16, 32, 0, stream>>>(feat, Wp1f, bp1, Wp2f, bp2,
                                                     Wv1f, bv1, Wv2, bv2, probs, value);
}
